// PointerGenOutputDecoder_61125974556736
// MI455X (gfx1250) — compile-verified, matrix operand order checked by reading, not run on hardware
//
#include <hip/hip_runtime.h>
#include <hip/hip_bf16.h>
#include <stdint.h>

// Pointer-generator decoder for MI455X (gfx1250, wave32).
// bf16 WMMA (v_wmma_f32_16x16x32_bf16) GEMMs with f32 accumulate,
// 4x N-tile register blocking per wave (1 A-fragment feeds 4 WMMAs),
// 1024-thread single-workgroup LDS-resident LSTM scan,
// online-softmax vocab reduction (output is only 512 scalars).

#define B_ 8
#define T_ 64
#define S_ 256
#define D_ 512
#define V_ 50000
#define BT_ (B_*T_)
#define FD_ (4*D_)

typedef __attribute__((ext_vector_type(16))) __bf16 v16bf;
typedef __attribute__((ext_vector_type(8)))  __bf16 v8bf;
typedef __attribute__((ext_vector_type(8)))  float  v8f;

__device__ __forceinline__ float sigm(float x) { return 1.f / (1.f + __expf(-x)); }

// A/B fragment for v_wmma_f32_16x16x32_bf16 (wave32):
// lane L holds row (L&15); K chunks: [8*half .. +7] and [16+8*half .. +7].
__device__ __forceinline__ v16bf load_frag(const __bf16* rowp, int half) {
    const v8bf* q = (const v8bf*)(rowp + 8 * half);
    v8bf lo = q[0];
    v8bf hi = q[2];
    return __builtin_shufflevector(lo, hi, 0,1,2,3,4,5,6,7,8,9,10,11,12,13,14,15);
}

// C[m,n] = sum_k A[m,k] * W[n,k] (+bias[n]); optional bf16 mirror of C.
// Each wave owns 4 consecutive 16x16 C tiles along N (one A fragment -> 4 WMMAs).
// 8 waves (256 threads) per block => block covers 32 N-tiles.
__global__ __launch_bounds__(256)
void wmma_gemm_xwt(const __bf16* __restrict__ A, long lda, long sA,
                   const __bf16* __restrict__ W, long ldw, long sW,
                   const float* __restrict__ bias,
                   float* __restrict__ C, long ldc, long sC,
                   __bf16* __restrict__ Cbf, long ldcbf, long sCbf,
                   int M, int N, int K)
{
    const int lane = threadIdx.x & 31;
    const int wave = threadIdx.x >> 5;
    const int half = lane >> 4;
    const int l15  = lane & 15;
    const int nTile0 = (blockIdx.x * 8 + wave) * 4;
    const int mTile  = blockIdx.y;
    if (nTile0 * 16 >= N) return;
    const int z = blockIdx.z;

    int rowA = mTile * 16 + l15; if (rowA >= M) rowA = M - 1;
    const __bf16* ap = A + (long)z * sA + (long)rowA * lda;

    const __bf16* wp[4];
    #pragma unroll
    for (int j = 0; j < 4; ++j) {
        int rowW = (nTile0 + j) * 16 + l15; if (rowW >= N) rowW = N - 1;
        wp[j] = W + (long)z * sW + (long)rowW * ldw;
    }

    // gfx1250 global_prefetch_b8 on the streamed rows
    __builtin_prefetch(ap, 0, 1);
    __builtin_prefetch(wp[0], 0, 1);
    __builtin_prefetch(wp[2], 0, 1);

    v8f accs[4] = {};
    for (int k = 0; k < K; k += 32) {
        v16bf af = load_frag(ap + k, half);
        v16bf w0 = load_frag(wp[0] + k, half);
        v16bf w1 = load_frag(wp[1] + k, half);
        v16bf w2 = load_frag(wp[2] + k, half);
        v16bf w3 = load_frag(wp[3] + k, half);
        accs[0] = __builtin_amdgcn_wmma_f32_16x16x32_bf16(false, af, false, w0, (short)0, accs[0], false, false);
        accs[1] = __builtin_amdgcn_wmma_f32_16x16x32_bf16(false, af, false, w1, (short)0, accs[1], false, false);
        accs[2] = __builtin_amdgcn_wmma_f32_16x16x32_bf16(false, af, false, w2, (short)0, accs[2], false, false);
        accs[3] = __builtin_amdgcn_wmma_f32_16x16x32_bf16(false, af, false, w3, (short)0, accs[3], false, false);
    }
    #pragma unroll
    for (int j = 0; j < 4; ++j) {
        int n = (nTile0 + j) * 16 + l15;
        if (n >= N) continue;
        float bv = (bias != nullptr) ? bias[n] : 0.f;
        #pragma unroll
        for (int r = 0; r < 8; ++r) {
            int m = mTile * 16 + r + 8 * half;
            if (m < M) {
                float v = accs[j][r] + bv;
                if (C)   C[(long)z * sC + (long)m * ldc + n] = v;
                if (Cbf) Cbf[(long)z * sCbf + (long)m * ldcbf + n] = (__bf16)v;
            }
        }
    }
}

// Sequential LSTM scan: one 1024-thread workgroup (32 waves on one WGP),
// h/c/g resident in LDS (57 KB); recurrent h(8x512) @ W_hh^T(2048x512) via
// WMMA each step (4 tiles/wave/step).
__global__ __launch_bounds__(1024)
void lstm_scan(const float* __restrict__ gx, const __bf16* __restrict__ Whh,
               const float* __restrict__ b_ih, const float* __restrict__ b_hh,
               const float* __restrict__ h0, const float* __restrict__ c0,
               float* __restrict__ outh, __bf16* __restrict__ out_bf, long ld_obf)
{
    __shared__ __bf16 g_s[B_][FD_];   // 32 KB
    __shared__ float  c_s[B_][D_];    // 16 KB
    __shared__ __bf16 h_s[B_][D_];    //  8 KB
    const int tid  = threadIdx.x;
    const int lane = tid & 31, wave = tid >> 5;     // 32 waves
    const int half = lane >> 4, l15 = lane & 15;
    const int rA   = l15 & 7;         // pad M=8 -> 16: rows 8..15 duplicate, discarded

    for (int i = tid; i < B_ * D_; i += 1024) {
        int b = i >> 9, d = i & (D_ - 1);
        c_s[b][d] = c0[i];
        h_s[b][d] = (__bf16)h0[i];
    }
    __syncthreads();

    for (int t = 0; t < T_; ++t) {
        for (int nt = wave; nt < FD_ / 16; nt += 32) {
            const __bf16* wp = Whh + (long)(nt * 16 + l15) * D_;
            const __bf16* ap = &h_s[rA][0];
            v8f acc = {};
            for (int k = 0; k < D_; k += 32) {
                v16bf af = load_frag(ap + k, half);
                v16bf bf = load_frag(wp + k, half);
                acc = __builtin_amdgcn_wmma_f32_16x16x32_bf16(false, af, false, bf,
                                                              (short)0, acc, false, false);
            }
            int n = nt * 16 + l15;
            #pragma unroll
            for (int r = 0; r < 8; ++r) {
                int m = r + 8 * half;
                if (m < B_) g_s[m][n] = (__bf16)acc[r];
            }
        }
        __syncthreads();
        for (int i = tid; i < B_ * D_; i += 1024) {
            int b = i >> 9, d = i & (D_ - 1);
            long row = (long)b * T_ + t;
            const float* gxr = gx + row * FD_;
            float ig = (float)g_s[b][d]        + gxr[d]        + b_ih[d]        + b_hh[d];
            float fg = (float)g_s[b][D_+d]     + gxr[D_+d]     + b_ih[D_+d]     + b_hh[D_+d];
            float gg = (float)g_s[b][2*D_+d]   + gxr[2*D_+d]   + b_ih[2*D_+d]   + b_hh[2*D_+d];
            float og = (float)g_s[b][3*D_+d]   + gxr[3*D_+d]   + b_ih[3*D_+d]   + b_hh[3*D_+d];
            float c = sigm(fg) * c_s[b][d] + sigm(ig) * tanhf(gg);
            float h = sigm(og) * tanhf(c);
            c_s[b][d] = c;
            h_s[b][d] = (__bf16)h;
            outh[row * D_ + d] = h;
            out_bf[row * ld_obf + d] = (__bf16)h;
        }
        __syncthreads();
    }
}

__global__ void k_f32_to_bf16(const float* __restrict__ s, __bf16* __restrict__ d, long n) {
    long i  = (long)blockIdx.x * blockDim.x + threadIdx.x;
    long st = (long)gridDim.x * blockDim.x;
    for (; i < n; i += st) d[i] = (__bf16)s[i];
}

__global__ void k_embed(const int* __restrict__ words, const float* __restrict__ tab,
                        float* __restrict__ emb, __bf16* __restrict__ emb_bf) {
    long i  = (long)blockIdx.x * blockDim.x + threadIdx.x;
    long st = (long)gridDim.x * blockDim.x;
    for (; i < (long)BT_ * D_; i += st) {
        int row = (int)(i >> 9), d = (int)(i & (D_ - 1));
        float v = tab[(long)words[row] * D_ + d];
        emb[i] = v;
        emb_bf[i] = (__bf16)v;
    }
}

// enc (B,S,D) -> enc_bf (B*S,D) and enc_t_bf (B,D,S)
__global__ void k_enc_prep(const float* __restrict__ enc, __bf16* __restrict__ enc_bf,
                           __bf16* __restrict__ enc_t_bf) {
    long i  = (long)blockIdx.x * blockDim.x + threadIdx.x;
    long st = (long)gridDim.x * blockDim.x;
    for (; i < (long)B_ * S_ * D_; i += st) {
        int b = (int)(i >> 17);
        int r = (int)(i & (S_ * D_ - 1));
        int s = r >> 9, d = r & (D_ - 1);
        float v = enc[i];
        enc_bf[i] = (__bf16)v;
        enc_t_bf[(long)b * D_ * S_ + (long)d * S_ + s] = (__bf16)v;
    }
}

// e[b,s,t] = sum_d tanh(wh[b,s,d] + ws[b,t,d]) * vt[d]
__global__ __launch_bounds__(256)
void k_attn_e(const float* __restrict__ wh, const float* __restrict__ ws,
              const float* __restrict__ vt, float* __restrict__ e)
{
    __shared__ float wh_s[D_], vt_s[D_];
    const int bs = blockIdx.x;            // b*S + s
    const int b = bs >> 8, s = bs & (S_ - 1);
    const float* whr = wh + (long)bs * D_;
    for (int i = threadIdx.x; i < D_; i += 256) { wh_s[i] = whr[i]; vt_s[i] = vt[i]; }
    __syncthreads();
    const int lane = threadIdx.x & 31, wave = threadIdx.x >> 5;
    for (int t = wave; t < T_; t += 8) {
        const float* wsr = ws + ((long)b * T_ + t) * D_;
        float p = 0.f;
        for (int d = lane; d < D_; d += 32)
            p += tanhf(wh_s[d] + wsr[d]) * vt_s[d];
        for (int m = 16; m >= 1; m >>= 1) p += __shfl_xor(p, m, 32);
        if (lane == 0) e[(long)b * S_ * T_ + (long)s * T_ + t] = p;
    }
}

// softmax over S for each (b,t); also emit transposed bf16 a_t (B*T, S)
__global__ __launch_bounds__(256)
void k_softmax_s(const float* __restrict__ e, float* __restrict__ a,
                 __bf16* __restrict__ a_t_bf)
{
    __shared__ float red[256];
    const int bt = blockIdx.x;
    const int b = bt >> 6, t = bt & (T_ - 1);
    const int s = threadIdx.x;            // S_ == 256
    const long idx = (long)b * S_ * T_ + (long)s * T_ + t;
    float v = e[idx];
    red[s] = v; __syncthreads();
    for (int off = 128; off >= 1; off >>= 1) {
        if (s < off) red[s] = fmaxf(red[s], red[s + off]);
        __syncthreads();
    }
    float mx = red[0]; __syncthreads();
    float ex = __expf(v - mx);
    red[s] = ex; __syncthreads();
    for (int off = 128; off >= 1; off >>= 1) {
        if (s < off) red[s] += red[s + off];
        __syncthreads();
    }
    float av = ex / red[0];
    a[idx] = av;
    a_t_bf[(long)bt * S_ + s] = (__bf16)av;
}

// per (b,t): softmax prob of target token, single-pass online softmax stats
__global__ __launch_bounds__(256)
void k_vocab_sel(const float* __restrict__ logits, const int* __restrict__ words,
                 float* __restrict__ sel)
{
    __shared__ float redm[256], reds[256];
    const int row = blockIdx.x;
    const float* lr = logits + (long)row * V_;
    float m = -3.4e38f, s = 0.f;
    for (int i = threadIdx.x; i < V_; i += 256) {
        float v = lr[i];
        float mn = fmaxf(m, v);
        s = s * __expf(m - mn) + __expf(v - mn);
        m = mn;
    }
    redm[threadIdx.x] = m; reds[threadIdx.x] = s; __syncthreads();
    for (int off = 128; off >= 1; off >>= 1) {
        if (threadIdx.x < off) {
            float m1 = redm[threadIdx.x], s1 = reds[threadIdx.x];
            float m2 = redm[threadIdx.x + off], s2 = reds[threadIdx.x + off];
            float mn = fmaxf(m1, m2);
            redm[threadIdx.x] = mn;
            reds[threadIdx.x] = s1 * __expf(m1 - mn) + s2 * __expf(m2 - mn);
        }
        __syncthreads();
    }
    if (threadIdx.x == 0) {
        int b = row >> 6, t = row & (T_ - 1);
        int tgt = words[b * T_ + ((t < T_ - 1) ? t + 1 : T_ - 1)];
        sel[row] = __expf(lr[tgt] - redm[0]) / reds[0];
    }
}

// p_gen + pointer mixture -> final 512 outputs
__global__ __launch_bounds__(256)
void k_final(const float* __restrict__ context, const float* __restrict__ outh,
             const float* __restrict__ emb, const float* __restrict__ Whp,
             const float* __restrict__ Wst, const float* __restrict__ Wst_b,
             const float* __restrict__ Wxt, const float* __restrict__ a,
             const int* __restrict__ words, const int* __restrict__ src,
             const float* __restrict__ sel, float* __restrict__ outp)
{
    __shared__ float red[256];
    const int row = blockIdx.x;
    const int b = row >> 6, t = row & (T_ - 1);
    float p = 0.f;
    for (int d = threadIdx.x; d < D_; d += 256) {
        long i = (long)row * D_ + d;
        p += context[i] * Whp[d] + outh[i] * Wst[d] + emb[i] * Wxt[d];
    }
    red[threadIdx.x] = p; __syncthreads();
    for (int off = 128; off >= 1; off >>= 1) {
        if (threadIdx.x < off) red[threadIdx.x] += red[threadIdx.x + off];
        __syncthreads();
    }
    float pg = sigm(red[0] + Wst_b[0]);
    __syncthreads();
    int tgt = words[b * T_ + ((t < T_ - 1) ? t + 1 : T_ - 1)];
    int s = threadIdx.x;                  // S_ == 256
    float m = (src[b * S_ + s] == tgt) ? a[(long)b * S_ * T_ + (long)s * T_ + t] : 0.f;
    red[s] = m; __syncthreads();
    for (int off = 128; off >= 1; off >>= 1) {
        if (s < off) red[s] += red[s + off];
        __syncthreads();
    }
    if (threadIdx.x == 0) outp[row] = pg * sel[row] + (1.f - pg) * red[0];
}

extern "C" void kernel_launch(void* const* d_in, const int* in_sizes, int n_in,
                              void* d_out, int out_size, void* d_ws, size_t ws_size,
                              hipStream_t stream)
{
    const int*   words = (const int*)  d_in[0];
    const float* enc   = (const float*)d_in[2];
    const float* pre_h = (const float*)d_in[3];
    const float* cell  = (const float*)d_in[4];
    const int*   src   = (const int*)  d_in[5];
    const float* embT  = (const float*)d_in[7];
    const float* W_ih  = (const float*)d_in[8];
    const float* W_hh  = (const float*)d_in[9];
    const float* b_ih  = (const float*)d_in[10];
    const float* b_hh  = (const float*)d_in[11];
    const float* Wh    = (const float*)d_in[12];
    const float* Ws_w  = (const float*)d_in[13];
    const float* Ws_b  = (const float*)d_in[14];
    const float* vt    = (const float*)d_in[15];
    const float* Whp   = (const float*)d_in[16];
    const float* Wst_w = (const float*)d_in[17];
    const float* Wst_b = (const float*)d_in[18];
    const float* Wxt   = (const float*)d_in[19];
    const float* V_w   = (const float*)d_in[20];
    const float* V_b   = (const float*)d_in[21];
    const float* Vp_w  = (const float*)d_in[22];
    const float* Vp_b  = (const float*)d_in[23];

    char* p = (char*)d_ws;
    auto alloc = [&](size_t bytes) -> char* {
        char* r = p; p += (bytes + 255) & ~(size_t)255; return r;
    };
    float*  emb       = (float*) alloc((size_t)BT_ * D_ * 4);
    __bf16* emb_bf    = (__bf16*)alloc((size_t)BT_ * D_ * 2);
    __bf16* Wih_bf    = (__bf16*)alloc((size_t)FD_ * D_ * 2);
    __bf16* Whh_bf    = (__bf16*)alloc((size_t)FD_ * D_ * 2);
    __bf16* Wh_bf     = (__bf16*)alloc((size_t)D_ * D_ * 2);
    __bf16* Ws_bf     = (__bf16*)alloc((size_t)D_ * D_ * 2);
    __bf16* Vw_bf     = (__bf16*)alloc((size_t)D_ * 2 * D_ * 2);
    __bf16* Vpw_bf    = (__bf16*)alloc((size_t)V_ * D_ * 2);
    float*  gx        = (float*) alloc((size_t)BT_ * FD_ * 4);
    float*  outh      = (float*) alloc((size_t)BT_ * D_ * 4);
    float*  wh        = (float*) alloc((size_t)B_ * S_ * D_ * 4);
    float*  wsb       = (float*) alloc((size_t)BT_ * D_ * 4);
    float*  e_buf     = (float*) alloc((size_t)B_ * S_ * T_ * 4);
    float*  a_buf     = (float*) alloc((size_t)B_ * S_ * T_ * 4);
    __bf16* a_t_bf    = (__bf16*)alloc((size_t)BT_ * S_ * 2);
    __bf16* enc_bf    = (__bf16*)alloc((size_t)B_ * S_ * D_ * 2);
    __bf16* enc_t_bf  = (__bf16*)alloc((size_t)B_ * D_ * S_ * 2);
    float*  context   = (float*) alloc((size_t)BT_ * D_ * 4);
    __bf16* hid_in_bf = (__bf16*)alloc((size_t)BT_ * 2 * D_ * 2);  // [context | out]
    float*  hid       = (float*) alloc((size_t)BT_ * D_ * 4);
    __bf16* hid_bf    = (__bf16*)alloc((size_t)BT_ * D_ * 2);
    float*  logits    = (float*) alloc((size_t)BT_ * V_ * 4);
    float*  sel       = (float*) alloc((size_t)BT_ * 4);

    // weight conversions
    k_f32_to_bf16<<<2048, 256, 0, stream>>>(W_ih, Wih_bf, (long)FD_ * D_);
    k_f32_to_bf16<<<2048, 256, 0, stream>>>(W_hh, Whh_bf, (long)FD_ * D_);
    k_f32_to_bf16<<<1024, 256, 0, stream>>>(Wh,   Wh_bf,  (long)D_ * D_);
    k_f32_to_bf16<<<1024, 256, 0, stream>>>(Ws_w, Ws_bf,  (long)D_ * D_);
    k_f32_to_bf16<<<1024, 256, 0, stream>>>(V_w,  Vw_bf,  (long)D_ * 2 * D_);
    k_f32_to_bf16<<<16384, 256, 0, stream>>>(Vp_w, Vpw_bf, (long)V_ * D_);

    k_embed<<<1024, 256, 0, stream>>>(words, embT, emb, emb_bf);
    k_enc_prep<<<2048, 256, 0, stream>>>(enc, enc_bf, enc_t_bf);

    // gx = emb @ W_ih^T (512 x 2048 x 512); 128 N-tiles -> x=4
    wmma_gemm_xwt<<<dim3(4, 32, 1), 256, 0, stream>>>(
        emb_bf, D_, 0, Wih_bf, D_, 0, nullptr,
        gx, FD_, 0, nullptr, 0, 0, BT_, FD_, D_);

    // LSTM scan (writes outh f32 and bf16 into hid_in_bf[:,512:1024])
    lstm_scan<<<1, 1024, 0, stream>>>(gx, Whh_bf, b_ih, b_hh, pre_h, cell,
                                      outh, hid_in_bf + D_, 2 * D_);

    // wh = enc @ Wh^T (2048 x 512 x 512); 32 N-tiles -> x=1
    wmma_gemm_xwt<<<dim3(1, 128, 1), 256, 0, stream>>>(
        enc_bf, D_, 0, Wh_bf, D_, 0, nullptr,
        wh, D_, 0, nullptr, 0, 0, B_ * S_, D_, D_);

    // ws = out @ Ws^T + Ws_b (512 x 512 x 512)
    wmma_gemm_xwt<<<dim3(1, 32, 1), 256, 0, stream>>>(
        hid_in_bf + D_, 2 * D_, 0, Ws_bf, D_, 0, Ws_b,
        wsb, D_, 0, nullptr, 0, 0, BT_, D_, D_);

    k_attn_e<<<B_ * S_, 256, 0, stream>>>(wh, wsb, vt, e_buf);
    k_softmax_s<<<BT_, 256, 0, stream>>>(e_buf, a_buf, a_t_bf);

    // context[b] = a_t[b] (64x256) @ enc_t[b]^T (512x256), batched over b
    wmma_gemm_xwt<<<dim3(1, 4, B_), 256, 0, stream>>>(
        a_t_bf, S_, (long)T_ * S_, enc_t_bf, S_, (long)D_ * S_, nullptr,
        context, D_, (long)T_ * D_, hid_in_bf, 2 * D_, (long)T_ * 2 * D_,
        T_, D_, S_);

    // hid = [context|out] @ V_w^T + V_b (512 x 512 x 1024)
    wmma_gemm_xwt<<<dim3(1, 32, 1), 256, 0, stream>>>(
        hid_in_bf, 2 * D_, 0, Vw_bf, 2 * D_, 0, V_b,
        hid, D_, 0, hid_bf, D_, 0, BT_, D_, 2 * D_);

    // logits = hid @ Vp_w^T + Vp_b (512 x 50000 x 512) -- dominant GEMM
    // 3125 N-tiles / 32 per block -> x=98
    wmma_gemm_xwt<<<dim3(98, 32, 1), 256, 0, stream>>>(
        hid_bf, D_, 0, Vpw_bf, D_, 0, Vp_b,
        logits, V_, 0, nullptr, 0, 0, BT_, V_, D_);

    k_vocab_sel<<<BT_, 256, 0, stream>>>(logits, words, sel);
    k_final<<<BT_, 256, 0, stream>>>(context, outh, emb, Whp, Wst_w, Wst_b, Wxt,
                                     a_buf, words, src, sel, (float*)d_out);
}
